// PlainSelfLayer_13348758356150
// MI455X (gfx1250) — matrix operand order repaired and checked by reading, not compiled, not run on hardware
//
#include <hip/hip_runtime.h>

// ---------------------------------------------------------------------------
// PlainSelfLayer for MI455X (gfx1250, wave32, WMMA).
// All GEMMs run as split-bf16 (hi+lo, 3 products) on v_wmma_f32_16x16x32_bf16
// with f32 accumulation (~fp32 accuracy at 3x bf16 cost, ~8x faster per issue
// than the 16x16x4 f32 WMMA path).
// ---------------------------------------------------------------------------

typedef __attribute__((ext_vector_type(16))) __bf16          v16bf;
typedef __attribute__((ext_vector_type(8)))  float           v8f;
typedef __attribute__((ext_vector_type(4)))  float           f32x4;
typedef __attribute__((ext_vector_type(8)))  unsigned short  u16x8;
typedef __attribute__((ext_vector_type(4)))  unsigned short  u16x4;

union FragBF { v16bf v; u16x8 h[2]; };

__device__ __forceinline__ void split_bf16(float x, unsigned short& hi, unsigned short& lo) {
    unsigned u = __float_as_uint(x);
    hi = (unsigned short)(u >> 16);                 // truncate-to-bf16 (exact residual)
    float r = x - __uint_as_float(u & 0xffff0000u); // residual
    lo = (unsigned short)(__float_as_uint(r) >> 16);
}

#define TB 256
constexpr int LDW = 40;   // LDS row stride in shorts: 16B-aligned, conflict-breaking

// out[m][n] = sum_k A[m][k] * B(k,n)   (A always K-contiguous)
//   BK=false: B[n*ldb + k] (row n is K-contiguous; used for W rows and Qp rows)
//   BK=true : B[k*ldb + n] (natural KxN; used for attn @ WV)
// MODE 0: plain   MODE 1: + bias[n]   MODE 2: *scale + mask[n*maskld+m]*maskval
template<int MODE, bool BK>
__global__ __launch_bounds__(TB) void gemm_bf16x3(
    const float* __restrict__ Ag, long lda, long strideA,
    const float* __restrict__ Bg, long ldb, long strideB,
    float*       __restrict__ Cg, long ldc, long strideC,
    int Kdim,
    const float* __restrict__ bias,
    const int*   __restrict__ maskg, long maskld, long strideMask,
    float scale, float maskval)
{
    __shared__ __align__(16) unsigned short As_h[128 * LDW];
    __shared__ __align__(16) unsigned short As_l[128 * LDW];
    __shared__ __align__(16) unsigned short Bs_h[128 * LDW];
    __shared__ __align__(16) unsigned short Bs_l[128 * LDW];

    const int  t      = threadIdx.x;
    const long bz     = blockIdx.z;
    const int  tile_n = blockIdx.x * 128;
    const int  tile_m = blockIdx.y * 128;

    const float* A = Ag + bz * strideA;
    const float* B = Bg + bz * strideB;
    float*       C = Cg + bz * strideC;

    const int w    = t >> 5;          // wave 0..7
    const int lane = t & 31;
    const int wm   = (w & 3) * 32;    // wave M offset in 128-tile
    const int wn   = (w >> 2) * 64;   // wave N offset in 128-tile
    const int lr   = lane & 15;
    const int lh   = lane >> 4;

    v8f acc[2][4];
    #pragma unroll
    for (int mi = 0; mi < 2; ++mi)
        #pragma unroll
        for (int ni = 0; ni < 4; ++ni)
            #pragma unroll
            for (int r = 0; r < 8; ++r) acc[mi][ni][r] = 0.0f;

    for (int k0 = 0; k0 < Kdim; k0 += 32) {
        // ---- stage A tile [128 x 32] as hi/lo bf16 ----
        #pragma unroll
        for (int i = 0; i < 4; ++i) {
            int lin = t + i * TB;          // 0..1023
            int m   = lin >> 3;            // row in tile
            int kg  = (lin & 7) << 2;      // k group of 4
            f32x4 vv = *(const f32x4*)(A + (long)(tile_m + m) * lda + k0 + kg);
            u16x4 hh, ll;
            #pragma unroll
            for (int j = 0; j < 4; ++j) {
                unsigned short h, l; split_bf16(vv[j], h, l);
                hh[j] = h; ll[j] = l;
            }
            *(u16x4*)&As_h[m * LDW + kg] = hh;
            *(u16x4*)&As_l[m * LDW + kg] = ll;
        }
        // ---- stage B tile into LDS as [n][k] hi/lo bf16 ----
        if (!BK) {
            #pragma unroll
            for (int i = 0; i < 4; ++i) {
                int lin = t + i * TB;
                int n   = lin >> 3;
                int kg  = (lin & 7) << 2;
                f32x4 vv = *(const f32x4*)(B + (long)(tile_n + n) * ldb + k0 + kg);
                u16x4 hh, ll;
                #pragma unroll
                for (int j = 0; j < 4; ++j) {
                    unsigned short h, l; split_bf16(vv[j], h, l);
                    hh[j] = h; ll[j] = l;
                }
                *(u16x4*)&Bs_h[n * LDW + kg] = hh;
                *(u16x4*)&Bs_l[n * LDW + kg] = ll;
            }
        } else {
            // B is [k][n]: transpose while staging
            #pragma unroll
            for (int i = 0; i < 4; ++i) {
                int lin = t + i * TB;
                int k   = lin >> 5;            // 0..31
                int ng  = (lin & 31) << 2;     // n group of 4
                f32x4 vv = *(const f32x4*)(B + (long)(k0 + k) * ldb + tile_n + ng);
                #pragma unroll
                for (int j = 0; j < 4; ++j) {
                    unsigned short h, l; split_bf16(vv[j], h, l);
                    Bs_h[(ng + j) * LDW + k] = h;
                    Bs_l[(ng + j) * LDW + k] = l;
                }
            }
        }
        __syncthreads();

        // ---- fragment loads (documented 16-bit WMMA VGPR layouts) ----
        FragBF ah[2], al[2], bh[4], bl[4];
        #pragma unroll
        for (int mi = 0; mi < 2; ++mi) {
            int base = (wm + mi * 16 + lr) * LDW + lh * 8; // A: K chunks lh*8 and 16+lh*8
            ah[mi].h[0] = *(const u16x8*)&As_h[base];
            ah[mi].h[1] = *(const u16x8*)&As_h[base + 16];
            al[mi].h[0] = *(const u16x8*)&As_l[base];
            al[mi].h[1] = *(const u16x8*)&As_l[base + 16];
        }
        #pragma unroll
        for (int ni = 0; ni < 4; ++ni) {
            int base = (wn + ni * 16 + lr) * LDW + lh * 8; // B: K chunks lh*8 and 16+lh*8
            bh[ni].h[0] = *(const u16x8*)&Bs_h[base];
            bh[ni].h[1] = *(const u16x8*)&Bs_h[base + 16];
            bl[ni].h[0] = *(const u16x8*)&Bs_l[base];
            bl[ni].h[1] = *(const u16x8*)&Bs_l[base + 16];
        }

        // ---- 2x4 tiles x 3 split products = 24 WMMAs per K-step ----
        #pragma unroll
        for (int mi = 0; mi < 2; ++mi)
            #pragma unroll
            for (int ni = 0; ni < 4; ++ni) {
                v8f c = acc[mi][ni];
                c = __builtin_amdgcn_wmma_f32_16x16x32_bf16(false, ah[mi].v, false, bh[ni].v, (short)0, c, false, false);
                c = __builtin_amdgcn_wmma_f32_16x16x32_bf16(false, ah[mi].v, false, bl[ni].v, (short)0, c, false, false);
                c = __builtin_amdgcn_wmma_f32_16x16x32_bf16(false, al[mi].v, false, bh[ni].v, (short)0, c, false, false);
                acc[mi][ni] = c;
            }
        __syncthreads();
    }

    // ---- epilogue: C/D layout lanes0-15 -> M=r, lanes16-31 -> M=8+r; N=lr ----
    #pragma unroll
    for (int mi = 0; mi < 2; ++mi)
        #pragma unroll
        for (int ni = 0; ni < 4; ++ni)
            #pragma unroll
            for (int r = 0; r < 8; ++r) {
                long mg = tile_m + wm + mi * 16 + lh * 8 + r;
                long ng = tile_n + wn + ni * 16 + lr;
                float val = acc[mi][ni][r];
                if (MODE == 1) val += bias[ng];
                if (MODE == 2) val = val * scale
                                   + (float)maskg[bz * strideMask + ng * maskld + mg] * maskval;
                C[mg * ldc + ng] = val;
            }
}

// Row softmax over contiguous rows (scores already stored transposed: [b][k][q])
__global__ __launch_bounds__(TB) void softmax_rows_q(float* __restrict__ data, int n)
{
    __shared__ float red[TB];
    float* row = data + (long)blockIdx.x * n;
    const int t = threadIdx.x;

    float mx = -3.4e38f;
    for (int i = t; i < n; i += TB) mx = fmaxf(mx, row[i]);
    red[t] = mx; __syncthreads();
    for (int s = TB / 2; s > 0; s >>= 1) {
        if (t < s) red[t] = fmaxf(red[t], red[t + s]);
        __syncthreads();
    }
    mx = red[0]; __syncthreads();

    float sum = 0.0f;
    for (int i = t; i < n; i += TB) {
        float e = expf(row[i] - mx);
        row[i] = e;
        sum += e;
    }
    red[t] = sum; __syncthreads();
    for (int s = TB / 2; s > 0; s >>= 1) {
        if (t < s) red[t] += red[t + s];
        __syncthreads();
    }
    float inv = 1.0f / red[0];
    for (int i = t; i < n; i += TB) row[i] *= inv;
}

extern "C" void kernel_launch(void* const* d_in, const int* in_sizes, int n_in,
                              void* d_out, int out_size, void* d_ws, size_t ws_size,
                              hipStream_t stream)
{
    constexpr long Bb = 8, Ss = 2048, Dd = 1024, Ah = 512, Dv = 512;

    const float* Q    = (const float*)d_in[0];
    const float* Kin  = (const float*)d_in[1];
    const float* V    = (const float*)d_in[2];
    const int*   mask = (const int*)  d_in[3];
    const float* Wq   = (const float*)d_in[4];
    const float* bq   = (const float*)d_in[5];
    const float* Wk   = (const float*)d_in[6];
    const float* bk   = (const float*)d_in[7];
    const float* Wv   = (const float*)d_in[8];
    const float* bv   = (const float*)d_in[9];

    float* out     = (float*)d_out;
    float* selfOut = out;                         // [B,S,Dv]
    float* attn    = out + (size_t)(Bb * Ss * Dv); // [B,S,S]

    // workspace: Qp, Kp, WV projections (f32), 3 * 32 MB = 96 MB
    float* Qp = (float*)d_ws;
    float* Kp = Qp + (size_t)(Bb * Ss * Ah);
    float* WV = Kp + (size_t)(Bb * Ss * Ah);

    const float scale = 0.04419417382415922f;   // 1/sqrt(512)
    dim3 blk(TB);

    // --- projections: M = B*S = 16384, N = 512, K = 1024, B rows = W rows ---
    dim3 gproj((unsigned)(Ah / 128), (unsigned)((Bb * Ss) / 128), 1);
    gemm_bf16x3<1, false><<<gproj, blk, 0, stream>>>(
        Q,   Dd, 0, Wq, Dd, 0, Qp, Ah, 0, (int)Dd, bq, nullptr, 0, 0, 1.0f, 0.0f);
    gemm_bf16x3<1, false><<<gproj, blk, 0, stream>>>(
        Kin, Dd, 0, Wk, Dd, 0, Kp, Ah, 0, (int)Dd, bk, nullptr, 0, 0, 1.0f, 0.0f);
    gemm_bf16x3<1, false><<<gproj, blk, 0, stream>>>(
        V,   Dd, 0, Wv, Dd, 0, WV, Dv, 0, (int)Dd, bv, nullptr, 0, 0, 1.0f, 0.0f);

    // --- transposed scores: St[b][k][q] = scale*Kp[k].Qp[q] + mask[b][q][k]*(-1e9) ---
    dim3 gsc((unsigned)(Ss / 128), (unsigned)(Ss / 128), (unsigned)Bb);
    gemm_bf16x3<2, false><<<gsc, blk, 0, stream>>>(
        Kp, Ah, Ss * Ah, Qp, Ah, Ss * Ah, attn, Ss, Ss * Ss,
        (int)Ah, nullptr, mask, Ss, Ss * Ss, scale, -1e9f);

    // --- softmax over q (contiguous rows of St) -> attn, in place in d_out ---
    softmax_rows_q<<<dim3((unsigned)(Bb * Ss)), blk, 0, stream>>>(attn, (int)Ss);

    // --- selfOutput[b][k][v] = sum_q attn[b][k][q] * WV[b][q][v]  (B is K-major) ---
    dim3 gout((unsigned)(Dv / 128), (unsigned)(Ss / 128), (unsigned)Bb);
    gemm_bf16x3<0, true><<<gout, blk, 0, stream>>>(
        attn, Ss, Ss * Ss, WV, Dv, Ss * Dv, selfOut, Dv, Ss * Dv,
        (int)Ss, nullptr, nullptr, 0, 0, 1.0f, 0.0f);
}
